// MultiHeadAttention_2594160246919
// MI455X (gfx1250) — matrix operand order repaired and checked by reading, not compiled, not run on hardware
//
#include <hip/hip_runtime.h>
#include <math.h>

#define D_MODEL 512
#define NUM_HEADS 8
#define D_HEAD 64
#define BATCH 2
#define SEQ 2048

typedef __attribute__((ext_vector_type(16))) __bf16 v16bf;
typedef __attribute__((ext_vector_type(8)))  float  v8f;
typedef __attribute__((ext_vector_type(4)))  unsigned int u32x4;

union Frag128 {
  v16bf v;
  u32x4 q[2];
};

__device__ __forceinline__ unsigned short f32_to_bf16(float f) {
  unsigned int u = __float_as_uint(f);
  u += 0x7FFFu + ((u >> 16) & 1u);   // round-to-nearest-even
  return (unsigned short)(u >> 16);
}

__device__ __forceinline__ v8f vzero8() {
  v8f v;
#pragma unroll
  for (int i = 0; i < 8; ++i) v[i] = 0.0f;
  return v;
}

__device__ __forceinline__ v8f wmma_bf16(v16bf a, v16bf b, v8f c) {
  return __builtin_amdgcn_wmma_f32_16x16x32_bf16(
      false, a, false, b, (short)0, c, false, false);
}

// A fragment 16x32 bf16 from row-major [row][col] (stride mult of 8 ushorts).
// lanes 0-15: row=lane, VGPR0-3 K=0..7, VGPR4-7 K=16..23; lanes16-31: K=8..15/24..31.
__device__ __forceinline__ v16bf load_a_frag(const unsigned short* base, int stride, int lane) {
  const int hf = lane >> 4, r = lane & 15;
  const unsigned short* row = base + r * stride;
  Frag128 f;
  f.q[0] = *(const u32x4*)(row + hf * 8);         // K = hf*8 .. +7
  f.q[1] = *(const u32x4*)(row + 16 + hf * 8);    // K = 16+hf*8 .. +7
  return f.v;
}

// B fragment 32x16 where source is [n][k] row-major (k contiguous per n-row).
// lanes 0-15: col n=lane, K=0..7,16..23; lanes 16-31: K=8..15,24..31.
__device__ __forceinline__ v16bf load_bt_frag(const unsigned short* base, int stride, int lane) {
  const int hf = lane >> 4, r = lane & 15;
  const unsigned short* row = base + r * stride + hf * 8;
  Frag128 f;
  f.q[0] = *(const u32x4*)(row);                  // K = hf*8 .. +7
  f.q[1] = *(const u32x4*)(row + 16);             // K = 16+hf*8 .. +7
  return f.v;
}

// ---------------------------------------------------------------- converts
__global__ void cvt_f32_bf16_kernel(const float* __restrict__ src,
                                    unsigned short* __restrict__ dst, int n) {
  int i = blockIdx.x * blockDim.x + threadIdx.x;
  int stride = gridDim.x * blockDim.x;
  for (; i < n; i += stride) dst[i] = f32_to_bf16(src[i]);
}

// 2x2-block transposed store helper: src row-major [k][ncols], dst [n][kstride]
__device__ __forceinline__ void store_t_2x2(const unsigned short* src, int ncols,
                                            unsigned short* dst, int dstride,
                                            int kb, int nb) {
  unsigned int u0 = *(const unsigned int*)(src + (size_t)(2 * kb) * ncols + 2 * nb);
  unsigned int u1 = *(const unsigned int*)(src + (size_t)(2 * kb + 1) * ncols + 2 * nb);
  unsigned int lo = (u0 & 0xFFFFu) | (u1 << 16);            // (k,n),(k+1,n)
  unsigned int hi = (u0 >> 16) | (u1 & 0xFFFF0000u);        // (k,n+1),(k+1,n+1)
  *(unsigned int*)(dst + (2 * nb) * dstride + 2 * kb) = lo;
  *(unsigned int*)(dst + (2 * nb + 1) * dstride + 2 * kb) = hi;
}

// ---------------------------------------------------------------- QKV projection
#define GA_STRIDE 40   // A tile row stride (ushorts), 80B = mult of 16
#define GW_STRIDE 40   // transposed W tile row stride

__global__ __launch_bounds__(256) void qkv_proj_kernel(
    const unsigned short* __restrict__ xb,   // [B, S, 512] bf16
    const unsigned short* __restrict__ wb,   // [3, H, 512, 64] bf16
    const float* __restrict__ bq, const float* __restrict__ bk,
    const float* __restrict__ bv,
    unsigned short* __restrict__ qkv)        // [3, B*H, S, 64] bf16
{
  __shared__ __align__(16) unsigned short ldsA[256 * GA_STRIDE];   // 256x32 tile
  __shared__ __align__(16) unsigned short ldsWt[64 * GW_STRIDE];   // [n][k] 64x32

  const int tid = threadIdx.x, lane = tid & 31, wave = tid >> 5;
  const int z = blockIdx.y;
  const int which = z / (BATCH * NUM_HEADS);
  const int rem = z % (BATCH * NUM_HEADS);
  const int b = rem / NUM_HEADS, h = rem % NUM_HEADS;
  const int m_base = blockIdx.x * 256;

  const unsigned short* A = xb + (size_t)b * SEQ * D_MODEL;
  const unsigned short* W = wb + ((size_t)which * NUM_HEADS + h) * D_MODEL * D_HEAD;
  const float* bias = (which == 0 ? bq : (which == 1 ? bk : bv)) + h * D_HEAD;

  v8f acc[8];
#pragma unroll
  for (int t = 0; t < 8; ++t) acc[t] = vzero8();

  for (int k0 = 0; k0 < D_MODEL; k0 += 32) {
    __syncthreads();
    // A tile 256x32: 512 u32x4 copies, 2 per thread
#pragma unroll
    for (int it = 0; it < 2; ++it) {
      int idx = tid + it * 256;
      int rowi = idx >> 1, qi = idx & 1;
      *(u32x4*)(&ldsA[rowi * GA_STRIDE + qi * 16]) =
          *(const u32x4*)(A + (size_t)(m_base + rowi) * D_MODEL + k0 + qi * 16);
    }
    // W tile 32x64 -> transposed [64][32]: 512 2x2 blocks, 2 per thread
#pragma unroll
    for (int it = 0; it < 2; ++it) {
      int idx = tid + it * 256;
      int nb = idx & 31, kb = idx >> 5;                 // kb 0..15, nb 0..31
      store_t_2x2(W + (size_t)k0 * D_HEAD, D_HEAD, ldsWt, GW_STRIDE, kb, nb);
    }
    __syncthreads();

    v16bf a0 = load_a_frag(&ldsA[(wave * 32) * GA_STRIDE], GA_STRIDE, lane);
    v16bf a1 = load_a_frag(&ldsA[(wave * 32 + 16) * GA_STRIDE], GA_STRIDE, lane);
#pragma unroll
    for (int t = 0; t < 4; ++t) {
      v16bf bf = load_bt_frag(&ldsWt[(t * 16) * GW_STRIDE], GW_STRIDE, lane);
      acc[t]     = wmma_bf16(a0, bf, acc[t]);
      acc[4 + t] = wmma_bf16(a1, bf, acc[4 + t]);
    }
  }

  const int hf = lane >> 4, r = lane & 15;
  unsigned short* out = qkv + (size_t)z * SEQ * D_HEAD;
#pragma unroll
  for (int t = 0; t < 4; ++t) {
    float bcol = bias[t * 16 + r];
#pragma unroll
    for (int i = 0; i < 8; ++i) {
      int row0 = m_base + wave * 32 + i + 8 * hf;
      out[(size_t)row0 * D_HEAD + t * 16 + r]        = f32_to_bf16(acc[t][i] + bcol);
      out[(size_t)(row0 + 16) * D_HEAD + t * 16 + r] = f32_to_bf16(acc[4 + t][i] + bcol);
    }
  }
}

// ---------------------------------------------------------------- flash attention
#define KV_STRIDE 72   // 144B rows, mult of 16
#define P_STRIDE  72

__global__ __launch_bounds__(256) void attn_kernel(
    const unsigned short* __restrict__ qkv,  // [3, B*H, S, 64] bf16
    unsigned short* __restrict__ ob)         // [B, S, 512] bf16 (concat order)
{
  __shared__ __align__(16) unsigned short ldsK[64 * KV_STRIDE];    // [n][e]
  __shared__ __align__(16) unsigned short ldsVt[64 * KV_STRIDE];   // [n][k] transposed
  __shared__ __align__(16) unsigned short ldsP[8 * 16 * P_STRIDE];

  const int tid = threadIdx.x, lane = tid & 31, wave = tid >> 5;
  const int hf = lane >> 4, r = lane & 15;
  const int bh = blockIdx.y;
  const int b = bh / NUM_HEADS, h = bh % NUM_HEADS;
  const int m_base = blockIdx.x * 128;

  const size_t plane = (size_t)SEQ * D_HEAD;
  const unsigned short* Q = qkv + (size_t)bh * plane;
  const unsigned short* K = qkv + (size_t)(BATCH * NUM_HEADS + bh) * plane;
  const unsigned short* V = qkv + (size_t)(2 * BATCH * NUM_HEADS + bh) * plane;

  // Q fragments (16x64 per wave) in registers, loaded straight from global.
  const unsigned short* qrow = Q + (size_t)(m_base + wave * 16) * D_HEAD;
  v16bf aq0 = load_a_frag(qrow, D_HEAD, lane);
  v16bf aq1 = load_a_frag(qrow + 32, D_HEAD, lane);

  v8f acc[4];
  float run_m[8], run_l[8];
#pragma unroll
  for (int t = 0; t < 4; ++t) acc[t] = vzero8();
#pragma unroll
  for (int i = 0; i < 8; ++i) { run_m[i] = -1e30f; run_l[i] = 0.0f; }

  const float scale = 0.125f;
  unsigned short* Pw = &ldsP[wave * 16 * P_STRIDE];

  for (int kt = 0; kt < SEQ; kt += 64) {
    __syncthreads();
    // K tile 64x64 direct copy: 256 u32x4, 1 per thread
    {
      int rowi = tid >> 2, qi = tid & 3;
      *(u32x4*)(&ldsK[rowi * KV_STRIDE + qi * 16]) =
          *(const u32x4*)(K + (size_t)(kt + rowi) * D_HEAD + qi * 16);
    }
    // V tile 64x64 transposed: 1024 2x2 blocks, 4 per thread
#pragma unroll
    for (int it = 0; it < 4; ++it) {
      int idx = tid + it * 256;
      int nb = idx & 31, kb = idx >> 5;                // kb 0..31, nb 0..31
      store_t_2x2(V + (size_t)kt * D_HEAD, D_HEAD, ldsVt, KV_STRIDE, kb, nb);
    }
    // prefetch next K/V tiles into cache while we compute
    if (kt + 64 < SEQ) {
      __builtin_prefetch(K + (size_t)(kt + 64 + (tid & 63)) * D_HEAD, 0, 1);
      __builtin_prefetch(V + (size_t)(kt + 64 + (tid & 63)) * D_HEAD, 0, 1);
    }
    __syncthreads();

    // S = Q K^T   (16 x 64 per wave)
    v8f sc[4];
#pragma unroll
    for (int t = 0; t < 4; ++t) {
      v8f c = vzero8();
      c = wmma_bf16(aq0, load_bt_frag(&ldsK[(t * 16) * KV_STRIDE], KV_STRIDE, lane), c);
      c = wmma_bf16(aq1, load_bt_frag(&ldsK[(t * 16) * KV_STRIDE + 32], KV_STRIDE, lane), c);
#pragma unroll
      for (int i = 0; i < 8; ++i) c[i] *= scale;
      sc[t] = c;
    }

    // online softmax; row i+8*hf is striped across the lane's 16-lane group
    float nm[8], alpha[8];
#pragma unroll
    for (int i = 0; i < 8; ++i) {
      float m = sc[0][i];
#pragma unroll
      for (int t = 1; t < 4; ++t) m = fmaxf(m, sc[t][i]);
#pragma unroll
      for (int off = 8; off >= 1; off >>= 1)
        m = fmaxf(m, __shfl_xor(m, off, 16));
      nm[i] = fmaxf(run_m[i], m);
      alpha[i] = __expf(run_m[i] - nm[i]);
      run_m[i] = nm[i];
    }
#pragma unroll
    for (int t = 0; t < 4; ++t)
#pragma unroll
      for (int i = 0; i < 8; ++i)
        sc[t][i] = __expf(sc[t][i] - nm[i]);
#pragma unroll
    for (int i = 0; i < 8; ++i) {
      float s = sc[0][i] + sc[1][i] + sc[2][i] + sc[3][i];
#pragma unroll
      for (int off = 8; off >= 1; off >>= 1)
        s += __shfl_xor(s, off, 16);
      run_l[i] = run_l[i] * alpha[i] + s;
    }
#pragma unroll
    for (int t = 0; t < 4; ++t)
#pragma unroll
      for (int i = 0; i < 8; ++i)
        acc[t][i] *= alpha[i];

    // P (bf16) -> per-wave LDS tile (C-layout -> A-layout via LDS)
#pragma unroll
    for (int t = 0; t < 4; ++t)
#pragma unroll
      for (int i = 0; i < 8; ++i)
        Pw[(i + 8 * hf) * P_STRIDE + t * 16 + r] = f32_to_bf16(sc[t][i]);

    __syncthreads();

    // O += P V   (V fragments contiguous thanks to transposed store)
    v16bf pa0 = load_a_frag(Pw, P_STRIDE, lane);
    v16bf pa1 = load_a_frag(Pw + 32, P_STRIDE, lane);
#pragma unroll
    for (int t = 0; t < 4; ++t) {
      acc[t] = wmma_bf16(pa0, load_bt_frag(&ldsVt[(t * 16) * KV_STRIDE], KV_STRIDE, lane), acc[t]);
      acc[t] = wmma_bf16(pa1, load_bt_frag(&ldsVt[(t * 16) * KV_STRIDE + 32], KV_STRIDE, lane), acc[t]);
    }
  }

#pragma unroll
  for (int i = 0; i < 8; ++i) run_l[i] = 1.0f / run_l[i];
#pragma unroll
  for (int t = 0; t < 4; ++t)
#pragma unroll
    for (int i = 0; i < 8; ++i) {
      int grow = m_base + wave * 16 + i + 8 * hf;
      int col = h * D_HEAD + t * 16 + r;
      ob[((size_t)b * SEQ + grow) * D_MODEL + col] = f32_to_bf16(acc[t][i] * run_l[i]);
    }
}

// ---------------------------------------------------------------- output projection
__global__ __launch_bounds__(256) void out_proj_kernel(
    const unsigned short* __restrict__ ob,   // [B*S, 512] bf16
    const unsigned short* __restrict__ wob,  // [512, 512] bf16
    const float* __restrict__ bo,
    float* __restrict__ out)                 // [B*S, 512] f32
{
  __shared__ __align__(16) unsigned short ldsA[256 * GA_STRIDE];
  __shared__ __align__(16) unsigned short ldsWt[64 * GW_STRIDE];

  const int tid = threadIdx.x, lane = tid & 31, wave = tid >> 5;
  const int m_base = blockIdx.x * 256;
  const int n_base = blockIdx.y * 64;

  v8f acc[8];
#pragma unroll
  for (int t = 0; t < 8; ++t) acc[t] = vzero8();

  for (int k0 = 0; k0 < D_MODEL; k0 += 32) {
    __syncthreads();
#pragma unroll
    for (int it = 0; it < 2; ++it) {
      int idx = tid + it * 256;
      int rowi = idx >> 1, qi = idx & 1;
      *(u32x4*)(&ldsA[rowi * GA_STRIDE + qi * 16]) =
          *(const u32x4*)(ob + (size_t)(m_base + rowi) * D_MODEL + k0 + qi * 16);
    }
#pragma unroll
    for (int it = 0; it < 2; ++it) {
      int idx = tid + it * 256;
      int nb = idx & 31, kb = idx >> 5;
      store_t_2x2(wob + (size_t)k0 * D_MODEL + n_base, D_MODEL, ldsWt, GW_STRIDE, kb, nb);
    }
    __syncthreads();

    v16bf a0 = load_a_frag(&ldsA[(wave * 32) * GA_STRIDE], GA_STRIDE, lane);
    v16bf a1 = load_a_frag(&ldsA[(wave * 32 + 16) * GA_STRIDE], GA_STRIDE, lane);
#pragma unroll
    for (int t = 0; t < 4; ++t) {
      v16bf bf = load_bt_frag(&ldsWt[(t * 16) * GW_STRIDE], GW_STRIDE, lane);
      acc[t]     = wmma_bf16(a0, bf, acc[t]);
      acc[4 + t] = wmma_bf16(a1, bf, acc[4 + t]);
    }
  }

  const int hf = lane >> 4, r = lane & 15;
#pragma unroll
  for (int t = 0; t < 4; ++t) {
    float bcol = bo[n_base + t * 16 + r];
#pragma unroll
    for (int i = 0; i < 8; ++i) {
      int row0 = m_base + wave * 32 + i + 8 * hf;
      out[(size_t)row0 * D_MODEL + n_base + t * 16 + r]        = acc[t][i] + bcol;
      out[(size_t)(row0 + 16) * D_MODEL + n_base + t * 16 + r] = acc[4 + t][i] + bcol;
    }
  }
}

// ---------------------------------------------------------------- launch
extern "C" void kernel_launch(void* const* d_in, const int* in_sizes, int n_in,
                              void* d_out, int out_size, void* d_ws, size_t ws_size,
                              hipStream_t stream) {
  (void)in_sizes; (void)n_in; (void)out_size; (void)ws_size;
  const float* x  = (const float*)d_in[0];
  const float* Wq = (const float*)d_in[1];
  const float* Wk = (const float*)d_in[2];
  const float* Wv = (const float*)d_in[3];
  const float* bq = (const float*)d_in[4];
  const float* bk = (const float*)d_in[5];
  const float* bv = (const float*)d_in[6];
  const float* Wo = (const float*)d_in[7];
  const float* bo = (const float*)d_in[8];

  unsigned short* ws = (unsigned short*)d_ws;
  const size_t N_X  = (size_t)BATCH * SEQ * D_MODEL;
  const size_t N_W1 = (size_t)NUM_HEADS * D_MODEL * D_HEAD;
  const size_t N_WO = (size_t)D_MODEL * D_MODEL;
  const size_t N_QKV= (size_t)3 * BATCH * NUM_HEADS * SEQ * D_HEAD;

  unsigned short* xb  = ws;
  unsigned short* wb  = xb + N_X;
  unsigned short* wob = wb + 3 * N_W1;
  unsigned short* qkv = wob + N_WO;
  unsigned short* obf = qkv + N_QKV;

  cvt_f32_bf16_kernel<<<512, 256, 0, stream>>>(x,  xb,  (int)N_X);
  cvt_f32_bf16_kernel<<<128, 256, 0, stream>>>(Wq, wb,             (int)N_W1);
  cvt_f32_bf16_kernel<<<128, 256, 0, stream>>>(Wk, wb + N_W1,      (int)N_W1);
  cvt_f32_bf16_kernel<<<128, 256, 0, stream>>>(Wv, wb + 2 * N_W1,  (int)N_W1);
  cvt_f32_bf16_kernel<<<128, 256, 0, stream>>>(Wo, wob,            (int)N_WO);

  qkv_proj_kernel<<<dim3(SEQ / 256, 3 * BATCH * NUM_HEADS), 256, 0, stream>>>(
      xb, wb, bq, bk, bv, qkv);

  attn_kernel<<<dim3(SEQ / 128, BATCH * NUM_HEADS), 256, 0, stream>>>(qkv, obf);

  out_proj_kernel<<<dim3(BATCH * SEQ / 256, D_MODEL / 64), 256, 0, stream>>>(
      obf, wob, bo, (float*)d_out);
}
